// TEMP_1692217115124
// MI455X (gfx1250) — matrix operand order repaired and checked by reading, not compiled, not run on hardware
//
#include <hip/hip_runtime.h>
#include <hip/hip_bf16.h>

// ---------------------------------------------------------------------------
// MI455X (gfx1250) fused tiny-attention:
//   bf16 WMMA (v_wmma_f32_16x16x32_bf16) for q-proj, kv-proj, out-proj.
//   One wave32 owns a 16-pixel tile; attention (H=16, dk=16, T=5) in VALU.
//   ~103 GFLOP GEMM / ~400MB HBM traffic -> bf16 WMMA puts compute at the
//   23.3 TB/s memory roofline; fp32 WMMA (K=4) would be ~8x compute-bound.
//   Round 2: register-cache invariant fragments (B-frags across t in kv-proj,
//   A-frags across n-tiles in q/out-proj) to cut redundant L2/LDS traffic.
// ---------------------------------------------------------------------------

typedef __attribute__((ext_vector_type(16))) __bf16 v16bf;
typedef __attribute__((ext_vector_type(8)))  float  v8f;

#define H_HEADS 16
#define DMODEL  256
#define SEQ     5
#define HW      16384   // 128*128 pixels per batch item

union FragBF { uint4 u[2]; v16bf v; };

// ------------------------- per-wave LDS layout (bytes) ----------------------
#define XB_ROW     264                       // 256 + 8 bf16 pad (bank-safe b128)
#define XB_OFF     0                         // bf16 [SEQ][16 pix][XB_ROW]
#define XB_BYTES   (SEQ * 16 * XB_ROW * 2)   // 42240
#define QB_OFF     (XB_OFF + XB_BYTES)       // bf16 [16 pix][256]
#define QB_BYTES   (16 * DMODEL * 2)         // 8192
#define OB_OFF     (QB_OFF + QB_BYTES)       // bf16 [16 pix][XB_ROW] attn out
#define OB_BYTES   (16 * XB_ROW * 2)         // 8448
#define KL_OFF     (OB_OFF + OB_BYTES)       // f32 [2 heads][SEQ][16 pix][17]
#define KL_BYTES   (2 * SEQ * 16 * 17 * 4)   // 10880
#define VL_OFF     (KL_OFF + KL_BYTES)       // f32 same shape
#define WAVE_BYTES (VL_OFF + KL_BYTES)       // 80640 -> 4 waves = 322560 <= 320KB
#define DT_OFF     KL_OFF                    // f32 [16 n][20] overlaps K (free then)

static __device__ __attribute__((always_inline))
v8f wmma_bf16(const FragBF& a, const FragBF& w, v8f c) {
  return __builtin_amdgcn_wmma_f32_16x16x32_bf16(
      /*neg_a=*/false, a.v, /*neg_b=*/false, w.v,
      /*c_mod=*/(short)0, c, /*reuse_a=*/false, /*reuse_b=*/false);
}

// 16-bit A layout: lane half h: VGPR0-3 = K 8h+0..7, VGPR4-7 = K 16+8h+0..7.
// row points at the lane's A row (LDS, row-major bf16, pixel = lane&15).
static __device__ __attribute__((always_inline))
FragBF load_a(const __bf16* row, int kt, int hl) {
  FragBF f;
  const __bf16* ap = row + kt * 32 + hl * 8;
  f.u[0] = *(const uint4*)ap;
  f.u[1] = *(const uint4*)(ap + 16);
  return f;
}

// 16-bit B layout: lane half h: VGPR0-3 = K 8h+0..7, VGPR4-7 = K 16+8h+0..7, column n = lane&15.
// row points at W^T[n] (global bf16, k-contiguous).
static __device__ __attribute__((always_inline))
FragBF load_w(const __bf16* row, int kt, int hl) {
  FragBF f;
  const __bf16* wp = row + kt * 32 + hl * 8;
  f.u[0] = *(const uint4*)wp;
  f.u[1] = *(const uint4*)(wp + 16);
  return f;
}

// --------------------- weight prep: fp32 W -> bf16 W^T ----------------------
// WtX[n][k] = W[k][n] (k contiguous) so B-fragments are 2x global_load_b128.
__global__ void prep_weights(const float* __restrict__ Wq,
                             const float* __restrict__ Wkv,
                             const float* __restrict__ Wp,
                             __bf16* __restrict__ WtQ,
                             __bf16* __restrict__ WtKV,
                             __bf16* __restrict__ WtP) {
  int n_all = blockIdx.x;      // 0..1023 : 256 (q) + 512 (kv) + 256 (p)
  int k     = threadIdx.x;     // 0..255
  if (n_all < 256) {
    WtQ[n_all * DMODEL + k] = (__bf16)Wq[k * 256 + n_all];
  } else if (n_all < 768) {
    int n = n_all - 256;
    WtKV[n * DMODEL + k] = (__bf16)Wkv[k * 512 + n];
  } else {
    int n = n_all - 768;
    WtP[n * DMODEL + k] = (__bf16)Wp[k * 256 + n];
  }
}

// ------------------------------ fused kernel --------------------------------
__global__ void __launch_bounds__(128, 1)
fused_attn(const float* __restrict__ x,
           const float* __restrict__ bq,
           const float* __restrict__ bkv,
           const float* __restrict__ bp,
           const float* __restrict__ pos_bias,
           const __bf16* __restrict__ WtQ,
           const __bf16* __restrict__ WtKV,
           const __bf16* __restrict__ WtP,
           float* __restrict__ out) {
  extern __shared__ __attribute__((aligned(16))) char smem[];
  const int lane = threadIdx.x & 31;
  const int wave = threadIdx.x >> 5;
  const int b    = blockIdx.x >> 8;                       // 256 WGs per batch
  const int p0   = (blockIdx.x & 255) * 64 + wave * 16;   // this wave's pixels

  char*   base = smem + wave * WAVE_BYTES;   // per-wave private LDS, no barriers
  __bf16* xb = (__bf16*)(base + XB_OFF);
  __bf16* qb = (__bf16*)(base + QB_OFF);
  __bf16* ob = (__bf16*)(base + OB_OFF);
  float*  kl = (float*)(base + KL_OFF);
  float*  vl = (float*)(base + VL_OFF);
  float*  dt = (float*)(base + DT_OFF);

  const int pl = lane & 15;   // pixel within tile / N column of fragments
  const int hl = lane >> 4;   // lane half

  // ---- Stage x tile: fp32 [t][k][pixel-contig] -> bf16 LDS [t][pixel][k] ----
  for (int t = 0; t < SEQ; ++t) {
    const float* xt = x + ((size_t)(b * SEQ + t) * DMODEL) * HW + p0;
    for (int kb = 0; kb < DMODEL; kb += 4) {
      int k = kb + hl * 2;                       // two k rows per lane half
      float f0 = xt[(size_t)k * HW + pl];        // 16-lane coalesced rows
      float f1 = xt[(size_t)(k + 1) * HW + pl];
      __bf16* d = &xb[(t * 16 + pl) * XB_ROW + k];
      d[0] = (__bf16)f0;
      d[1] = (__bf16)f1;
    }
  }

  // ---- q = X[t=0] @ Wq + bq ----
  // A-fragments (t=0) are invariant across the 16 n-tiles: hold in registers.
  {
    FragBF ax[8];
#pragma unroll
    for (int kt = 0; kt < 8; ++kt) ax[kt] = load_a(&xb[pl * XB_ROW], kt, hl);
    for (int nt = 0; nt < 16; ++nt) {
      float bias = bq[nt * 16 + pl];
      v8f c0, c1;
#pragma unroll
      for (int i = 0; i < 8; ++i) { c0[i] = bias; c1[i] = 0.0f; }
      const __bf16* wrow = WtQ + (nt * 16 + pl) * DMODEL;
#pragma unroll
      for (int kt = 0; kt < 8; ++kt) {
        FragBF w = load_w(wrow, kt, hl);
        if (kt & 1) c1 = wmma_bf16(ax[kt], w, c1);
        else        c0 = wmma_bf16(ax[kt], w, c0);
      }
#pragma unroll
      for (int r = 0; r < 8; ++r)   // D: VGPR r -> pixel r+8h, lane%16 -> col
        qb[(r + 8 * hl) * DMODEL + nt * 16 + pl] = (__bf16)(c0[r] + c1[r]);
    }
  }

  // ---- per head-pair: K/V projection (WMMA) + softmax attention (VALU) ----
  for (int h0 = 0; h0 < H_HEADS; h0 += 2) {
#pragma unroll
    for (int j = 0; j < 4; ++j) {              // j<2: K cols, j>=2: V cols
      int hh   = j & 1;
      int ncol = ((j < 2) ? 0 : DMODEL) + (h0 + hh) * 16;
      float bias = bkv[ncol + pl];
      // Weight B-fragments are invariant across t: load once, reuse 5x.
      FragBF wf[8];
      const __bf16* wrow = WtKV + (ncol + pl) * DMODEL;
#pragma unroll
      for (int kt = 0; kt < 8; ++kt) wf[kt] = load_w(wrow, kt, hl);
      for (int t = 0; t < SEQ; ++t) {
        const __bf16* arow = &xb[(t * 16 + pl) * XB_ROW];
        v8f c0, c1;
#pragma unroll
        for (int i = 0; i < 8; ++i) { c0[i] = bias; c1[i] = 0.0f; }
#pragma unroll
        for (int kt = 0; kt < 8; ++kt) {
          FragBF a = load_a(arow, kt, hl);
          if (kt & 1) c1 = wmma_bf16(a, wf[kt], c1);
          else        c0 = wmma_bf16(a, wf[kt], c0);
        }
        float* dst = ((j < 2) ? kl : vl) + (hh * SEQ + t) * 16 * 17;
#pragma unroll
        for (int r = 0; r < 8; ++r)
          dst[(r + 8 * hl) * 17 + pl] = c0[r] + c1[r];  // [pixel][dk pad 17]
      }
    }
    // attention: lane owns (pixel=pl, head=h0+hl); scale = 1/temp = dk^0.5 = 4
    const int h = h0 + hl;
    float qv[16];
#pragma unroll
    for (int dk = 0; dk < 16; ++dk)
      qv[dk] = (float)qb[pl * DMODEL + h * 16 + dk];
    float s[SEQ];
#pragma unroll
    for (int t = 0; t < SEQ; ++t) {
      const float* kr = kl + ((hl * SEQ + t) * 16 + pl) * 17;
      float acc = 0.0f;
#pragma unroll
      for (int dk = 0; dk < 16; ++dk) acc += qv[dk] * kr[dk];
      s[t] = acc * 4.0f + pos_bias[h * SEQ + t];
    }
    float m = s[0];
#pragma unroll
    for (int t = 1; t < SEQ; ++t) m = fmaxf(m, s[t]);
    float sum = 0.0f;
#pragma unroll
    for (int t = 0; t < SEQ; ++t) { s[t] = __expf(s[t] - m); sum += s[t]; }
    float inv = 1.0f / sum;
#pragma unroll
    for (int dk = 0; dk < 16; ++dk) {
      float o = 0.0f;
#pragma unroll
      for (int t = 0; t < SEQ; ++t)
        o += s[t] * vl[((hl * SEQ + t) * 16 + pl) * 17 + dk];
      ob[pl * XB_ROW + h * 16 + dk] = (__bf16)(o * inv);
    }
  }

  // ---- out = attn_out @ Wp + bp, then coalesced float4 stores via LDS ----
  {
    FragBF ax[8];   // attn-out A-fragments invariant across n-tiles
#pragma unroll
    for (int kt = 0; kt < 8; ++kt) ax[kt] = load_a(&ob[pl * XB_ROW], kt, hl);
    for (int nt = 0; nt < 16; ++nt) {
      float bias = bp[nt * 16 + pl];
      v8f c0, c1;
#pragma unroll
      for (int i = 0; i < 8; ++i) { c0[i] = bias; c1[i] = 0.0f; }
      const __bf16* wrow = WtP + (nt * 16 + pl) * DMODEL;
#pragma unroll
      for (int kt = 0; kt < 8; ++kt) {
        FragBF w = load_w(wrow, kt, hl);
        if (kt & 1) c1 = wmma_bf16(ax[kt], w, c1);
        else        c0 = wmma_bf16(ax[kt], w, c0);
      }
#pragma unroll
      for (int r = 0; r < 8; ++r)                // transpose D -> [n][pixel]
        dt[pl * 20 + r + 8 * hl] = c0[r] + c1[r];
      int nn = lane >> 1, poff = (lane & 1) * 8; // 2 lanes per output row
      float4 v0 = *(const float4*)&dt[nn * 20 + poff];
      float4 v1 = *(const float4*)&dt[nn * 20 + poff + 4];
      float* g = out + ((size_t)(b * DMODEL + nt * 16 + nn)) * HW + p0 + poff;
      *(float4*)g       = v0;
      *(float4*)(g + 4) = v1;
    }
  }
}

// ------------------------------- launcher -----------------------------------
extern "C" void kernel_launch(void* const* d_in, const int* in_sizes, int n_in,
                              void* d_out, int out_size, void* d_ws, size_t ws_size,
                              hipStream_t stream) {
  (void)in_sizes; (void)n_in; (void)out_size; (void)ws_size;
  const float* x   = (const float*)d_in[0];
  const float* Wq  = (const float*)d_in[1];
  const float* bq  = (const float*)d_in[2];
  const float* Wkv = (const float*)d_in[3];
  const float* bkv = (const float*)d_in[4];
  const float* Wp  = (const float*)d_in[5];
  const float* bp  = (const float*)d_in[6];
  const float* pb  = (const float*)d_in[7];

  __bf16* WtQ  = (__bf16*)d_ws;                                  // 256x256 bf16
  __bf16* WtKV = (__bf16*)((char*)d_ws + 131072);                // 512x256 bf16
  __bf16* WtP  = (__bf16*)((char*)d_ws + 131072 + 262144);       // 256x256 bf16

  prep_weights<<<dim3(1024), dim3(256), 0, stream>>>(Wq, Wkv, Wp, WtQ, WtKV, WtP);

  // 4 batches x 256 tiles; 128 threads = 4 wave32, each wave owns 16 pixels.
  fused_attn<<<dim3(4 * 256), dim3(128), 4 * WAVE_BYTES, stream>>>(
      x, bq, bkv, bp, pb, WtQ, WtKV, WtP, (float*)d_out);
}
